// SWABlock_6528350290247
// MI455X (gfx1250) — matrix operand order repaired and checked by reading, not compiled, not run on hardware
//
#include <hip/hip_runtime.h>
#include <math.h>

// ---------------- problem constants ----------------
constexpr int T   = 2048;
constexpr int DIM = 1024;
constexpr int H   = 16;
constexpr int HD  = 64;
constexpr int WIN = 512;
constexpr int FFN = 4096;

typedef unsigned short us;
typedef __attribute__((ext_vector_type(4)))  unsigned int uint4v;
typedef __attribute__((ext_vector_type(4)))  float        float4v;
typedef __attribute__((ext_vector_type(16))) __bf16       v16bf;
typedef __attribute__((ext_vector_type(8)))  float        v8f;

// CDNA5 async memory->LDS path (gfx1250), used when the toolchain exposes it.
#if defined(__has_builtin)
#if __has_builtin(__builtin_amdgcn_global_load_async_to_lds_b128) && \
    __has_builtin(__builtin_amdgcn_s_wait_asynccnt)
#define USE_ASYNC_LDS 1
#endif
#endif
#ifndef USE_ASYNC_LDS
#define USE_ASYNC_LDS 0
#endif

#if USE_ASYNC_LDS
typedef int gv4i __attribute__((vector_size(4 * sizeof(int))));
typedef __attribute__((address_space(1))) gv4i* as1_v4i_ptr;   // global
typedef __attribute__((address_space(3))) gv4i* as3_v4i_ptr;   // LDS
#endif

union Frag {                 // 16 bf16 = 32 bytes = two 16B chunks
    uint4v u[2];
    v16bf  v;
};

static __device__ __forceinline__ us f2bf(float f) {
    unsigned int u = __float_as_uint(f);
    u += 0x7FFFu + ((u >> 16) & 1u);        // round-to-nearest-even
    return (us)(u >> 16);
}
static __device__ __forceinline__ float bf2f(us h) {
    return __uint_as_float(((unsigned int)h) << 16);
}
static __device__ __forceinline__ v8f wmma_bf16(v16bf a, v16bf b, v8f c) {
    return __builtin_amdgcn_wmma_f32_16x16x32_bf16(false, a, false, b, (short)0, c,
                                                   false, false);
}

// ---------------- fused RMSNorm -> bf16 ----------------
__global__ __launch_bounds__(256) void rmsnorm_kernel(const float* __restrict__ x,
                                                      const float* __restrict__ g,
                                                      us* __restrict__ out, int dim) {
    __shared__ float red[256];
    const int row = blockIdx.x;
    const int tid = threadIdx.x;
    const float* xr = x + (size_t)row * dim;
    float acc = 0.f;
    for (int i = tid; i < dim; i += 256) { float v = xr[i]; acc += v * v; }
    red[tid] = acc;
    __syncthreads();
    for (int s = 128; s > 0; s >>= 1) {
        if (tid < s) red[tid] += red[tid + s];
        __syncthreads();
    }
    const float inv = rsqrtf(red[0] / (float)dim + 1e-6f);
    us* orow = out + (size_t)row * dim;
    for (int i = tid; i < dim; i += 256) orow[i] = f2bf(xr[i] * inv * g[i]);
}

// ---------------- WMMA GEMM:  Y[M,N] = A[M,K](bf16) @ W[N,K](f32)^T ----------------
// MODE 0: Y fp32 (+optional residual)   MODE 1: Y bf16   MODE 2: Y bf16 transposed [N,M]
template <int MODE>
__global__ __launch_bounds__(256) void gemm_kernel(const us* __restrict__ A,
                                                   const float* __restrict__ W,
                                                   const float* __restrict__ res,
                                                   void* __restrict__ Y,
                                                   int M, int N, int K) {
    __shared__ __align__(16) us As[64][40];   // 64x32 bf16 tile (+pad, rows 16B aligned)
    __shared__ __align__(16) us Bs[64][40];   // Bs[n][k] = W[n0+n][k0+k] as bf16

    const int tid  = threadIdx.x;
    const int wave = tid >> 5;
    const int lane = tid & 31;
    const int half = lane >> 4;
    const int lr   = lane & 15;
    const int m0 = blockIdx.y * 64;
    const int n0 = blockIdx.x * 64;
    const int wm = wave & 3;        // 4 row tiles of 16
    const int wn = wave >> 2;       // 2 col strips of 32

    const int ldRow = tid >> 2;         // 0..63
    const int ldCol = (tid & 3) * 8;    // 0,8,16,24

    v8f acc[2] = {};

    for (int k0 = 0; k0 < K; k0 += 32) {
        __syncthreads();
        // stage A tile: memory -> LDS directly via the CDNA5 async mover when present
        const us* agp = A + (size_t)(m0 + ldRow) * K + k0 + ldCol;
#if USE_ASYNC_LDS
        __builtin_amdgcn_global_load_async_to_lds_b128(
            (as1_v4i_ptr)agp, (as3_v4i_ptr)&As[ldRow][ldCol], 0, 0);
#else
        *reinterpret_cast<uint4v*>(&As[ldRow][ldCol]) =
            *reinterpret_cast<const uint4v*>(agp);
#endif
        // stage W (fp32 -> bf16 convert in flight; weights stay L2-resident)
        {
            const float* wp = W + (size_t)(n0 + ldRow) * K + k0 + ldCol;
            float4v w0 = *reinterpret_cast<const float4v*>(wp);
            float4v w1 = *reinterpret_cast<const float4v*>(wp + 4);
            us* dst = &Bs[ldRow][ldCol];
            dst[0] = f2bf(w0.x); dst[1] = f2bf(w0.y);
            dst[2] = f2bf(w0.z); dst[3] = f2bf(w0.w);
            dst[4] = f2bf(w1.x); dst[5] = f2bf(w1.y);
            dst[6] = f2bf(w1.z); dst[7] = f2bf(w1.w);
        }
        if (k0 + 32 < K)
            __builtin_prefetch(W + (size_t)(n0 + ldRow) * K + k0 + 32 + ldCol, 0, 2);
#if USE_ASYNC_LDS
        __builtin_amdgcn_s_wait_asynccnt(0);
#endif
        __syncthreads();

        // A fragment (16x32): element e -> K = half*16 + e
        Frag a;
        a.u[0] = *reinterpret_cast<const uint4v*>(&As[wm * 16 + lr][half * 16]);
        a.u[1] = *reinterpret_cast<const uint4v*>(&As[wm * 16 + lr][half * 16 + 8]);

#pragma unroll
        for (int f = 0; f < 2; ++f) {
            // B fragment (32x16): element e -> K = half*16 + e ; lane col = lr
            const us* brow = &Bs[wn * 32 + f * 16 + lr][0];
            Frag b;
            b.u[0] = *reinterpret_cast<const uint4v*>(brow + half * 16);
            b.u[1] = *reinterpret_cast<const uint4v*>(brow + half * 16 + 8);
            acc[f] = wmma_bf16(a.v, b.v, acc[f]);
        }
    }

    // C/D layout: VGPR r -> row m = r + 8*half ; lane -> col lr
#pragma unroll
    for (int f = 0; f < 2; ++f) {
        const int ncol = n0 + wn * 32 + f * 16 + lr;
#pragma unroll
        for (int r = 0; r < 8; ++r) {
            const int mrow = m0 + wm * 16 + r + 8 * half;
            float v = acc[f][r];
            if (MODE == 0) {
                if (res) v += res[(size_t)mrow * N + ncol];
                reinterpret_cast<float*>(Y)[(size_t)mrow * N + ncol] = v;
            } else if (MODE == 1) {
                reinterpret_cast<us*>(Y)[(size_t)mrow * N + ncol] = f2bf(v);
            } else {
                reinterpret_cast<us*>(Y)[(size_t)ncol * M + mrow] = f2bf(v);
            }
        }
    }
}

// ---------------- sliding-window attention ----------------
// one wave per (head, 16-query tile). Q,K bf16 [T,DIM]; Vt bf16 [DIM,T]; out bf16 [T,DIM]
__global__ __launch_bounds__(32) void attn_kernel(const us* __restrict__ Qb,
                                                  const us* __restrict__ Kb,
                                                  const us* __restrict__ Vt,
                                                  us* __restrict__ Ob) {
    __shared__ __align__(16) float S[16][544];   // raw scores strip
    __shared__ __align__(16) us    P[16][544];   // softmax weights, bf16

    const int lane = threadIdx.x;
    const int half = lane >> 4;
    const int lr   = lane & 15;
    const int qt = blockIdx.x & (T / 16 - 1);
    const int h  = blockIdx.x >> 7;             // T/16 = 128 tiles per head
    const int q0 = qt * 16;

    int klo = q0 - (WIN - 1); if (klo < 0) klo = 0;
    const int t0     = klo >> 4;
    const int ntiles = qt - t0 + 1;             // <= 33
    const int ncols  = ntiles * 16;

    // Q fragments (A side, 16x32 each over hd 0..31 and 32..63)
    const us* qrow = Qb + (size_t)(q0 + lr) * DIM + h * HD;
    Frag qa0, qa1;
    qa0.u[0] = *reinterpret_cast<const uint4v*>(qrow + half * 16);
    qa0.u[1] = *reinterpret_cast<const uint4v*>(qrow + half * 16 + 8);
    qa1.u[0] = *reinterpret_cast<const uint4v*>(qrow + 32 + half * 16);
    qa1.u[1] = *reinterpret_cast<const uint4v*>(qrow + 32 + half * 16 + 8);

    // ---- phase 1: S = scale * Q K^T with causal+window mask ----
    for (int t = t0; t <= qt; ++t) {
        const int key = t * 16 + lr;            // this lane's key column
        const us* krow = Kb + (size_t)key * DIM + h * HD;
        Frag kb0, kb1;                           // B side: element e -> kdim = half*16+e
        kb0.u[0] = *reinterpret_cast<const uint4v*>(krow + half * 16);
        kb0.u[1] = *reinterpret_cast<const uint4v*>(krow + half * 16 + 8);
        kb1.u[0] = *reinterpret_cast<const uint4v*>(krow + 32 + half * 16);
        kb1.u[1] = *reinterpret_cast<const uint4v*>(krow + 32 + half * 16 + 8);
        v8f s = {};
        s = wmma_bf16(qa0.v, kb0.v, s);
        s = wmma_bf16(qa1.v, kb1.v, s);
        const int c = (t - t0) * 16 + lr;
#pragma unroll
        for (int r = 0; r < 8; ++r) {
            const int m = r + 8 * half;          // query row in tile
            const int i = q0 + m;
            const int j = t * 16 + lr;
            const bool ok = (j <= i) && (i - j < WIN);
            S[m][c] = ok ? s[r] * 0.125f : -1e9f;
        }
    }
    __syncthreads();

    // ---- phase 2: row softmax (lanes 0..15, one row each) ----
    if (lane < 16) {
        const int row = lane;
        float mx = -1e30f;
        for (int c = 0; c < ncols; ++c) mx = fmaxf(mx, S[row][c]);
        float sum = 0.f;
        for (int c = 0; c < ncols; ++c) {
            float e = __expf(S[row][c] - mx);
            S[row][c] = e;
            sum += e;
        }
        const float inv = 1.f / sum;
        for (int c = 0; c < ncols; ++c) P[row][c] = f2bf(S[row][c] * inv);
        const int pad = (ncols + 31) & ~31;
        for (int c = ncols; c < pad; ++c) P[row][c] = 0;
    }
    __syncthreads();

    // ---- phase 3: O = P @ V ----
    const int ncolsPad = (ncols + 31) & ~31;     // <= 544
    v8f o[4] = {};
    for (int kk = 0; kk < ncolsPad; kk += 32) {
        Frag pa;                                  // A side from LDS strip
        pa.u[0] = *reinterpret_cast<const uint4v*>(&P[lr][kk + half * 16]);
        pa.u[1] = *reinterpret_cast<const uint4v*>(&P[lr][kk + half * 16 + 8]);
        const int keyoff = t0 * 16 + kk + half * 16;
#pragma unroll
        for (int nt = 0; nt < 4; ++nt) {
            const us* vrow = Vt + (size_t)(h * HD + nt * 16 + lr) * T + keyoff;
            Frag vb;                              // B side: element e -> key = half*16+e
            vb.u[0] = *reinterpret_cast<const uint4v*>(vrow);
            vb.u[1] = *reinterpret_cast<const uint4v*>(vrow + 8);
            o[nt] = wmma_bf16(pa.v, vb.v, o[nt]);
        }
    }

#pragma unroll
    for (int nt = 0; nt < 4; ++nt) {
        const int col = h * HD + nt * 16 + lr;
#pragma unroll
        for (int r = 0; r < 8; ++r) {
            const int m = q0 + r + 8 * half;
            Ob[(size_t)m * DIM + col] = f2bf(o[nt][r]);
        }
    }
}

// ---------------- elementwise SwiGLU: out = silu(gate) * up (bf16) ----------------
__global__ __launch_bounds__(256) void silu_mul_kernel(const us* __restrict__ gate,
                                                       const us* __restrict__ up,
                                                       us* __restrict__ out, int n) {
    const int i = blockIdx.x * 256 + threadIdx.x;
    if (i < n) {
        const float g = bf2f(gate[i]);
        const float u = bf2f(up[i]);
        out[i] = f2bf((g / (1.f + __expf(-g))) * u);
    }
}

// ---------------- host-side orchestration ----------------
extern "C" void kernel_launch(void* const* d_in, const int* in_sizes, int n_in,
                              void* d_out, int out_size, void* d_ws, size_t ws_size,
                              hipStream_t stream) {
    (void)in_sizes; (void)n_in; (void)out_size; (void)ws_size;
    const float* x  = (const float*)d_in[0];
    const float* g1 = (const float*)d_in[1];
    const float* wq = (const float*)d_in[2];
    const float* wk = (const float*)d_in[3];
    const float* wv = (const float*)d_in[4];
    const float* wo = (const float*)d_in[5];
    const float* g2 = (const float*)d_in[6];
    const float* wg = (const float*)d_in[7];
    const float* wu = (const float*)d_in[8];
    const float* wd = (const float*)d_in[9];
    float* out = (float*)d_out;

    // scratch layout (~64 MB total, 256B aligned slices)
    char* base = (char*)d_ws;
    size_t off = 0;
    auto take = [&](size_t bytes) -> char* {
        char* p = base + off;
        off = (off + bytes + 255) & ~(size_t)255;
        return p;
    };
    us*    xnb   = (us*)take((size_t)T * DIM * 2);
    us*    Qb    = (us*)take((size_t)T * DIM * 2);
    us*    Kb    = (us*)take((size_t)T * DIM * 2);
    us*    Vt    = (us*)take((size_t)T * DIM * 2 + 4096);  // +pad: attn may over-read <=32B/row
    us*    attnb = (us*)take((size_t)T * DIM * 2);
    float* hbuf  = (float*)take((size_t)T * DIM * 4);
    us*    n2b   = (us*)take((size_t)T * DIM * 2);
    us*    gateb = (us*)take((size_t)T * FFN * 2);
    us*    upb   = (us*)take((size_t)T * FFN * 2);

    // 1) xn = rmsnorm(x)*g1  -> bf16
    rmsnorm_kernel<<<T, 256, 0, stream>>>(x, g1, xnb, DIM);

    // 2) Q,K (bf16 row-major), V (bf16 transposed [DIM,T])
    dim3 gQ(DIM / 64, T / 64);
    gemm_kernel<1><<<gQ, 256, 0, stream>>>(xnb, wq, nullptr, Qb, T, DIM, DIM);
    gemm_kernel<1><<<gQ, 256, 0, stream>>>(xnb, wk, nullptr, Kb, T, DIM, DIM);
    gemm_kernel<2><<<gQ, 256, 0, stream>>>(xnb, wv, nullptr, Vt, T, DIM, DIM);

    // 3) sliding-window attention -> bf16 [T,DIM]
    attn_kernel<<<H * (T / 16), 32, 0, stream>>>(Qb, Kb, Vt, attnb);

    // 4) h = x + attn @ wo^T   (fp32)
    gemm_kernel<0><<<gQ, 256, 0, stream>>>(attnb, wo, x, hbuf, T, DIM, DIM);

    // 5) n2 = rmsnorm(h)*g2 -> bf16
    rmsnorm_kernel<<<T, 256, 0, stream>>>(hbuf, g2, n2b, DIM);

    // 6) gate/up GEMMs -> bf16, then silu(gate)*up (in place into gateb)
    dim3 gF(FFN / 64, T / 64);
    gemm_kernel<1><<<gF, 256, 0, stream>>>(n2b, wg, nullptr, gateb, T, FFN, DIM);
    gemm_kernel<1><<<gF, 256, 0, stream>>>(n2b, wu, nullptr, upb, T, FFN, DIM);
    silu_mul_kernel<<<(T * FFN) / 256, 256, 0, stream>>>(gateb, upb, gateb, T * FFN);

    // 7) out = h + act @ wd^T  (fp32, final residual fused)
    gemm_kernel<0><<<gQ, 256, 0, stream>>>(gateb, wd, hbuf, out, T, DIM, FFN);
}
